// HybridAutoEncoder_65481071397456
// MI455X (gfx1250) — compile-verified, matrix operand order checked by reading, not run on hardware
//
#include <hip/hip_runtime.h>
#include <hip/hip_bf16.h>

typedef __attribute__((ext_vector_type(16))) __bf16 v16bf;
typedef __attribute__((ext_vector_type(8)))  float  v8f;

#define BM 256
#define BN 128
#define BK 32
#define LDTB 80   // bytes per LDS tile row: 64 data + 16 pad (keeps 16B chunks aligned)
#define LDTD 20   // dwords per LDS tile row

__device__ __forceinline__ unsigned int pack_bf16(float lo, float hi) {
    unsigned int a = __float_as_uint(lo);
    unsigned int b = __float_as_uint(hi);
    a += 0x7FFFu + ((a >> 16) & 1u);   // round-to-nearest-even
    b += 0x7FFFu + ((b >> 16) & 1u);
    return (a >> 16) | (b & 0xFFFF0000u);
}
__device__ __forceinline__ unsigned short f2bf(float f) {
    unsigned int u = __float_as_uint(f);
    u += 0x7FFFu + ((u >> 16) & 1u);
    return (unsigned short)(u >> 16);
}
__device__ __forceinline__ float bf2f(unsigned short u) {
    return __uint_as_float(((unsigned int)u) << 16);
}
// Low 32 bits of a flat shared-aperture pointer are the LDS byte offset.
__device__ __forceinline__ unsigned int lds_off(const void* p) {
    return (unsigned int)(unsigned long long)p;
}
__device__ __forceinline__ void async_b128(unsigned int lds, const void* g) {
    asm volatile("global_load_async_to_lds_b128 %0, %1, off"
                 :: "v"(lds), "v"((unsigned long long)g) : "memory");
}
__device__ __forceinline__ void wait_async0() {
    asm volatile("s_wait_asynccnt 0" ::: "memory");
}

// Stage one 256x32 A tile and one 128x32 W tile (bf16) into LDS via async copies.
__device__ __forceinline__ void issue_tile_loads(
    const unsigned short* __restrict__ A, const unsigned short* __restrict__ W,
    unsigned int asBase, unsigned int wsBase,
    int m0, int n0, int K, int k0, int tid)
{
    #pragma unroll
    for (int i = 0; i < 4; ++i) {               // A: 1024 chunks of 16B
        const int c = tid + i * 256;
        const int row = c >> 2, c16 = c & 3;
        async_b128(asBase + row * LDTB + c16 * 16,
                   A + (size_t)(m0 + row) * K + k0 + c16 * 8);
    }
    #pragma unroll
    for (int i = 0; i < 2; ++i) {               // W: 512 chunks of 16B
        const int c = tid + i * 256;
        const int row = c >> 2, c16 = c & 3;
        async_b128(wsBase + row * LDTB + c16 * 16,
                   W + (size_t)(n0 + row) * K + k0 + c16 * 8);
    }
}

// out[M,N] = act(A[M,K](bf16) @ W[N,K](bf16)^T + bias); M%256==0, N%128==0, K%32==0.
__global__ __launch_bounds__(256) void gemm_bf16_wmma(
    const unsigned short* __restrict__ A, const unsigned short* __restrict__ W,
    const float* __restrict__ bias, void* __restrict__ out,
    int M, int N, int K, int relu, int out_bf16)
{
    __shared__ __align__(16) unsigned int Ash[2][BM * LDTD];   // 2 x 20KB
    __shared__ __align__(16) unsigned int Wsh[2][BN * LDTD];   // 2 x 10KB

    const int tid  = threadIdx.x;
    const int lane = tid & 31;
    const int wave = tid >> 5;
    const int wm   = wave & 3;     // 4 waves x 64 rows = 256
    const int wn   = wave >> 2;    // 2 waves x 64 cols = 128
    const int m0   = blockIdx.y * BM;
    const int n0   = blockIdx.x * BN;
    const int r16  = lane & 15;
    const int half = lane >> 4;

    const unsigned int asB[2] = { lds_off(&Ash[0][0]), lds_off(&Ash[1][0]) };
    const unsigned int wsB[2] = { lds_off(&Wsh[0][0]), lds_off(&Wsh[1][0]) };

    v8f acc[4][4];
    #pragma unroll
    for (int mt = 0; mt < 4; ++mt)
        #pragma unroll
        for (int nt = 0; nt < 4; ++nt)
            acc[mt][nt] = (v8f)0.0f;

    const int steps = K / BK;
    issue_tile_loads(A, W, asB[0], wsB[0], m0, n0, K, 0, tid);

    for (int s = 0; s < steps; ++s) {
        wait_async0();        // this wave's async copies for buf[s&1] landed
        __syncthreads();      // all waves' copies landed; all waves done with buf[(s-1)&1]
        if (s + 1 < steps)
            issue_tile_loads(A, W, asB[(s + 1) & 1], wsB[(s + 1) & 1],
                             m0, n0, K, (s + 1) * BK, tid);

        const unsigned int* As = &Ash[s & 1][0];
        const unsigned int* Ws = &Wsh[s & 1][0];

        union Frag { v16bf v; unsigned int u[8]; };
        Frag af[4], bf[4];
        #pragma unroll
        for (int mt = 0; mt < 4; ++mt) {
            const int row = wm * 64 + mt * 16 + r16;
            #pragma unroll
            for (int j = 0; j < 8; ++j) {
                const int k = ((j >> 2) << 4) + half * 8 + ((j & 3) << 1);
                af[mt].u[j] = As[row * LDTD + (k >> 1)];
            }
        }
        #pragma unroll
        for (int nt = 0; nt < 4; ++nt) {
            const int row = wn * 64 + nt * 16 + r16;
            #pragma unroll
            for (int j = 0; j < 8; ++j) {
                const int k = ((j >> 2) << 4) + half * 8 + ((j & 3) << 1);
                bf[nt].u[j] = Ws[row * LDTD + (k >> 1)];
            }
        }
        #pragma unroll
        for (int mt = 0; mt < 4; ++mt)
            #pragma unroll
            for (int nt = 0; nt < 4; ++nt)
                acc[mt][nt] = __builtin_amdgcn_wmma_f32_16x16x32_bf16(
                    false, af[mt].v, false, bf[nt].v,
                    (short)0, acc[mt][nt], false, false);
    }

    // Epilogue. C layout: VGPR v -> M = v + 8*(lane/16), N = lane%16.
    #pragma unroll
    for (int mt = 0; mt < 4; ++mt) {
        #pragma unroll
        for (int nt = 0; nt < 4; ++nt) {
            const int gmBase = m0 + wm * 64 + mt * 16 + half * 8;
            const int gn     = n0 + wn * 64 + nt * 16 + r16;
            const float bv = bias[gn];
            #pragma unroll
            for (int v = 0; v < 8; ++v) {
                float r = acc[mt][nt][v] + bv;
                if (relu) r = fmaxf(r, 0.0f);
                if (out_bf16)
                    ((unsigned short*)out)[(size_t)(gmBase + v) * N + gn] = f2bf(r);
                else
                    ((float*)out)[(size_t)(gmBase + v) * N + gn] = r;
            }
        }
    }
}

// Elementwise f32 -> bf16 (4 elements per thread)
__global__ void cvt_f32_bf16(const float* __restrict__ in, unsigned int* __restrict__ out, int n4)
{
    const int i = blockIdx.x * blockDim.x + threadIdx.x;
    if (i >= n4) return;
    const float4 v = ((const float4*)in)[i];
    uint2 p;
    p.x = pack_bf16(v.x, v.y);
    p.y = pack_bf16(v.z, v.w);
    ((uint2*)out)[i] = p;
}

// z[b,o] = h2[b,:](bf16) . enc_w3[o,:] + b3[o]   (B=2048, N=12, K=1024)
__global__ void enc3_kernel(const unsigned short* __restrict__ A, const float* __restrict__ W3,
                            const float* __restrict__ b3, float* __restrict__ z)
{
    const int idx = blockIdx.x * blockDim.x + threadIdx.x;
    if (idx >= 2048 * 12) return;
    const int b = idx / 12, o = idx % 12;
    const unsigned short* a = A + (size_t)b * 1024;
    const float* w = W3 + (size_t)o * 1024;
    float s = 0.0f;
    for (int k = 0; k < 1024; ++k) s = fmaf(bf2f(a[k]), w[k], s);
    z[idx] = s + b3[o];
}

// g1[b,o] = relu(qz[b,:] . dec_w1[o,:] + b1[o]) as bf16   (B=2048, N=1024, K=12)
__global__ void dec1_kernel(const float* __restrict__ qz, const float* __restrict__ W,
                            const float* __restrict__ bias, unsigned short* __restrict__ out)
{
    const int idx = blockIdx.x * blockDim.x + threadIdx.x;
    if (idx >= 2048 * 1024) return;
    const int b = idx >> 10, o = idx & 1023;
    float s = 0.0f;
    #pragma unroll
    for (int k = 0; k < 12; ++k) s = fmaf(qz[b * 12 + k], W[o * 12 + k], s);
    s += bias[o];
    out[idx] = f2bf(fmaxf(s, 0.0f));
}

// 12-qubit state-vector simulation, one workgroup (128 threads) per batch item.
#define NQ 12
#define NSTATE 4096
#define QTHREADS 128

__global__ __launch_bounds__(QTHREADS) void quantum_latent_kernel(
    const float* __restrict__ z, const float* __restrict__ qp, float* __restrict__ qz)
{
    __shared__ float2 amp[NSTATE];
    __shared__ float  partial[NQ * QTHREADS];

    const int b   = blockIdx.x;
    const int tid = threadIdx.x;

    for (int i = tid; i < NSTATE; i += QTHREADS)
        amp[i] = (i == 0) ? make_float2(1.0f, 0.0f) : make_float2(0.0f, 0.0f);
    __syncthreads();

    for (int q = 0; q < NQ; ++q) {           // RX(theta_q), qubit q = bit (11-q)
        const float th = 0.5f * (z[(size_t)b * NQ + q] + qp[q]);
        const float c = cosf(th), s = sinf(th);
        const int bit = 11 - q;
        const int stride = 1 << bit;
        for (int p = tid; p < NSTATE / 2; p += QTHREADS) {
            const int i0 = ((p >> bit) << (bit + 1)) | (p & (stride - 1));
            const int i1 = i0 + stride;
            const float2 a0 = amp[i0], a1 = amp[i1];
            amp[i0] = make_float2(c * a0.x + s * a1.y, c * a0.y - s * a1.x);
            amp[i1] = make_float2(s * a0.y + c * a1.x, -s * a0.x + c * a1.y);
        }
        __syncthreads();
    }

    for (int rep = 0; rep < 2; ++rep) {      // CNOT rings: amplitude swaps
        for (int q = 0; q < NQ; ++q) {
            const int cmask = 1 << (11 - q);
            const int tmask = 1 << (11 - ((q + 1) % NQ));
            for (int i = tid; i < NSTATE; i += QTHREADS) {
                if ((i & cmask) && !(i & tmask)) {
                    const int j = i | tmask;
                    const float2 t0 = amp[i], t1 = amp[j];
                    amp[i] = t1; amp[j] = t0;
                }
            }
            __syncthreads();
        }
    }

    float acc[NQ];
    #pragma unroll
    for (int q = 0; q < NQ; ++q) acc[q] = 0.0f;
    for (int i = tid; i < NSTATE; i += QTHREADS) {
        const float2 a = amp[i];
        const float pr = a.x * a.x + a.y * a.y;
        #pragma unroll
        for (int q = 0; q < NQ; ++q)
            acc[q] += ((i >> (11 - q)) & 1) ? -pr : pr;
    }
    #pragma unroll
    for (int q = 0; q < NQ; ++q) partial[q * QTHREADS + tid] = acc[q];
    __syncthreads();
    if (tid < NQ) {
        float s = 0.0f;
        for (int t = 0; t < QTHREADS; ++t) s += partial[tid * QTHREADS + t];
        qz[(size_t)b * NQ + tid] = s;
    }
}

extern "C" void kernel_launch(void* const* d_in, const int* in_sizes, int n_in,
                              void* d_out, int out_size, void* d_ws, size_t ws_size,
                              hipStream_t stream)
{
    (void)in_sizes; (void)n_in; (void)out_size; (void)ws_size;
    const float* x       = (const float*)d_in[0];
    const float* enc_w1  = (const float*)d_in[1];
    const float* enc_b1  = (const float*)d_in[2];
    const float* enc_w2  = (const float*)d_in[3];
    const float* enc_b2  = (const float*)d_in[4];
    const float* enc_w3  = (const float*)d_in[5];
    const float* enc_b3  = (const float*)d_in[6];
    const float* qparams = (const float*)d_in[7];
    const float* dec_w1  = (const float*)d_in[8];
    const float* dec_b1  = (const float*)d_in[9];
    const float* dec_w2  = (const float*)d_in[10];
    const float* dec_b2  = (const float*)d_in[11];
    const float* dec_w3  = (const float*)d_in[12];
    const float* dec_b3  = (const float*)d_in[13];
    float* out = (float*)d_out;

    const int B = 2048, D = 4096, H1 = 2048, H2 = 1024, L = NQ;

    // workspace carve-up (256B aligned blocks)
    char* w = (char*)d_ws;
    auto carve = [&](size_t bytes) { void* p = (void*)w; w += (bytes + 255) & ~(size_t)255; return p; };
    unsigned short* xbf   = (unsigned short*)carve((size_t)B * D * 2);
    unsigned short* w1bf  = (unsigned short*)carve((size_t)H1 * D * 2);
    unsigned short* w2bf  = (unsigned short*)carve((size_t)H2 * H1 * 2);
    unsigned short* dw2bf = (unsigned short*)carve((size_t)H1 * H2 * 2);
    unsigned short* dw3bf = (unsigned short*)carve((size_t)D * H1 * 2);
    unsigned short* h1bf  = (unsigned short*)carve((size_t)B * H1 * 2);
    unsigned short* h2bf  = (unsigned short*)carve((size_t)B * H2 * 2);
    float*          zb    = (float*)carve((size_t)B * L * 4);
    float*          qzb   = (float*)carve((size_t)B * L * 4);
    unsigned short* g1bf  = (unsigned short*)carve((size_t)B * H2 * 2);
    unsigned short* g2bf  = (unsigned short*)carve((size_t)B * H1 * 2);

    // precision conversion (one-time per launch; pure bandwidth)
    auto cvt = [&](const float* src, unsigned short* dst, size_t n) {
        const int n4 = (int)(n / 4);
        cvt_f32_bf16<<<(n4 + 255) / 256, 256, 0, stream>>>(src, (unsigned int*)dst, n4);
    };
    cvt(x,      xbf,   (size_t)B * D);
    cvt(enc_w1, w1bf,  (size_t)H1 * D);
    cvt(enc_w2, w2bf,  (size_t)H2 * H1);
    cvt(dec_w2, dw2bf, (size_t)H1 * H2);
    cvt(dec_w3, dw3bf, (size_t)D * H1);

    // encoder
    gemm_bf16_wmma<<<dim3(H1 / BN, B / BM), 256, 0, stream>>>(xbf,  w1bf, enc_b1, h1bf, B, H1, D,  1, 1);
    gemm_bf16_wmma<<<dim3(H2 / BN, B / BM), 256, 0, stream>>>(h1bf, w2bf, enc_b2, h2bf, B, H2, H1, 1, 1);
    enc3_kernel<<<(B * L + 255) / 256, 256, 0, stream>>>(h2bf, enc_w3, enc_b3, zb);

    // quantum latent
    quantum_latent_kernel<<<B, QTHREADS, 0, stream>>>(zb, qparams, qzb);

    // decoder
    dec1_kernel<<<(B * H2 + 255) / 256, 256, 0, stream>>>(qzb, dec_w1, dec_b1, g1bf);
    gemm_bf16_wmma<<<dim3(H1 / BN, B / BM), 256, 0, stream>>>(g1bf, dw2bf, dec_b2, g2bf, B, H1, H2, 1, 1);
    gemm_bf16_wmma<<<dim3(D  / BN, B / BM), 256, 0, stream>>>(g2bf, dw3bf, dec_b3, out,  B, D,  H1, 0, 0);
}
